// q_PatchEmbed_30562987278755
// MI455X (gfx1250) — matrix operand order repaired and checked by reading, not compiled, not run on hardware
//
#include <hip/hip_runtime.h>
#include <hip/hip_bf16.h>

typedef __attribute__((ext_vector_type(16))) _Float16 v16h;
typedef __attribute__((ext_vector_type(8)))  _Float16 v8h;
typedef __attribute__((ext_vector_type(4)))  _Float16 v4h;
typedef __attribute__((ext_vector_type(8)))  float    v8f;

#define KDIM   768            // C*P*P
#define EDIM   768            // output channels
#define NPATCH 196            // 14*14
#define MTOT   25088          // B * NPATCH
#define MTILE  32             // block M tile (2 WMMA subtiles)
#define LDA    776            // padded LDS row stride in halves (avoids bank conflicts)
#define QMAXF  127.0f

// ---------------------------------------------------------------------------
// Kernel 1: per-out-channel int8 weight quantization (stored as f16 integers),
// conv_sf[e] = w_sf[e]*sf_in, fbias[e] = round(bias/conv_sf)*conv_sf
// ---------------------------------------------------------------------------
__global__ __launch_bounds__(256) void wquant_kernel(
    const float* __restrict__ w, const float* __restrict__ bias,
    const float* __restrict__ sf_ptr,
    _Float16* __restrict__ wq, float* __restrict__ conv_sf,
    float* __restrict__ fbias) {
  __shared__ float red[256];
  __shared__ float s_wsf;
  const int e = blockIdx.x;
  const int t = threadIdx.x;
  const float* we = w + (size_t)e * KDIM;
  float m = 0.0f;
  for (int k = t; k < KDIM; k += 256) m = fmaxf(m, fabsf(we[k]));
  red[t] = m;
  __syncthreads();
  for (int s = 128; s > 0; s >>= 1) {
    if (t < s) red[t] = fmaxf(red[t], red[t + s]);
    __syncthreads();
  }
  if (t == 0) {
    float wsf = red[0] * (1.0f / QMAXF);
    s_wsf = wsf;
    float csf = wsf * sf_ptr[0];
    conv_sf[e] = csf;
    fbias[e] = rintf(bias[e] / csf) * csf;
  }
  __syncthreads();
  const float inv = 1.0f / s_wsf;
  for (int k = t; k < KDIM; k += 256)
    wq[(size_t)e * KDIM + k] = (_Float16)rintf(we[k] * inv);
}

// ---------------------------------------------------------------------------
// Kernel 2: patchify + f16 WMMA GEMM (M=25088, K=768, N=768) + dequant epilogue
// + global activation absmax.
// Block = 32 patch rows x 768 cols; 8 waves, each wave 32 rows x 96 cols
// (2 M-subtiles x 6 N-tiles = 12 accumulators; each B fragment reused 2x).
// ---------------------------------------------------------------------------
__global__ __launch_bounds__(256) void gemm_kernel(
    const float* __restrict__ x, const _Float16* __restrict__ wq,
    const float* __restrict__ conv_sf, const float* __restrict__ fbias,
    const float* __restrict__ sf_ptr, float* __restrict__ out,
    unsigned* __restrict__ actmax) {
  __shared__ _Float16 Atile[MTILE * LDA];
  __shared__ float wmax[8];

  const int t  = threadIdx.x;
  const int m0 = blockIdx.x * MTILE;
  const float inv_sf = 1.0f / sf_ptr[0];

  // ---- patchify: load 32 rows x 768 K of x, scale by 1/sf, convert to f16 ----
  // 24576 elements as 6144 float4 segments; 256 threads x 24 iters.
#pragma unroll
  for (int it = 0; it < 24; ++it) {
    int v   = it * 256 + t;        // 0..6143
    int row = v / 192;             // 192 float4 per K-row
    int q   = v - row * 192;
    int d   = q * 4;               // K index (multiple of 4)
    int m   = m0 + row;
    int b   = m / NPATCH;
    int n   = m - b * NPATCH;
    int gh  = n / 14;
    int gw  = n - gh * 14;
    int c   = d >> 8;              // d / 256
    int rr  = (d & 255) >> 4;      // row inside patch
    int pw  = d & 15;              // col inside patch (mult of 4)
    const float4 xv = *(const float4*)(
        x + (((size_t)(b * 3 + c) * 224 + gh * 16 + rr) * 224 + gw * 16 + pw));
    v4h hv;
    hv.x = (_Float16)(xv.x * inv_sf);
    hv.y = (_Float16)(xv.y * inv_sf);
    hv.z = (_Float16)(xv.z * inv_sf);
    hv.w = (_Float16)(xv.w * inv_sf);
    *(v4h*)&Atile[row * LDA + d] = hv;
  }
  __syncthreads();

  const int wave  = t >> 5;
  const int lane  = t & 31;
  const int col16 = lane & 15;     // A row M (within subtile), B/C col N
  const int kgrp  = lane >> 4;     // lane half selects K sub-group
  const int e0    = wave * 96;     // 6 tiles of 16 cols per wave

  v8f acc0[6] = {};                // rows m0 + 0..15
  v8f acc1[6] = {};                // rows m0 + 16..31
  const _Float16* Arow0 = &Atile[col16 * LDA];
  const _Float16* Arow1 = &Atile[(16 + col16) * LDA];

  for (int ks = 0; ks < 24; ++ks) {
    const int kb = ks * 32;
    // A fragment 16x32 f16: lanes<16 hold K kb+0..15, lanes>=16 kb+16..31
    v8h a0lo = *(const v8h*)(Arow0 + kb + kgrp * 16);
    v8h a0hi = *(const v8h*)(Arow0 + kb + kgrp * 16 + 8);
    v8h a1lo = *(const v8h*)(Arow1 + kb + kgrp * 16);
    v8h a1hi = *(const v8h*)(Arow1 + kb + kgrp * 16 + 8);
    v16h a0 = __builtin_shufflevector(a0lo, a0hi, 0, 1, 2, 3, 4, 5, 6, 7, 8, 9,
                                      10, 11, 12, 13, 14, 15);
    v16h a1 = __builtin_shufflevector(a1lo, a1hi, 0, 1, 2, 3, 4, 5, 6, 7, 8, 9,
                                      10, 11, 12, 13, 14, 15);
#pragma unroll
    for (int tl = 0; tl < 6; ++tl) {
      const int e = e0 + tl * 16 + col16;
      // B fragment 32x16: lane col = lane%16, K = kb + (lane/16)*16 .. +15,
      // contiguous packed pairs -> direct 32B load from row-major wq[E][K];
      // reused by both M-subtiles (halves L2 traffic vs 16-row tiles).
      v16h bfrag = *(const v16h*)(wq + (size_t)e * KDIM + kb + kgrp * 16);
      acc0[tl] = __builtin_amdgcn_wmma_f32_16x16x32_f16(
          false, a0, false, bfrag, (short)0, acc0[tl], false, false);
      acc1[tl] = __builtin_amdgcn_wmma_f32_16x16x32_f16(
          false, a1, false, bfrag, (short)0, acc1[tl], false, false);
    }
  }

  // ---- epilogue: dequant + bias, store f32, track |out| max ----
  float lmax = 0.0f;
  const int rowbase = kgrp * 8;    // C/D layout: M = vgpr + 8*(lane/16)
#pragma unroll
  for (int tl = 0; tl < 6; ++tl) {
    const int e  = e0 + tl * 16 + col16;
    const float cs = conv_sf[e];
    const float fb = fbias[e];
#pragma unroll
    for (int r = 0; r < 8; ++r) {
      const int mA = m0 + rowbase + r;
      const int mB = m0 + 16 + rowbase + r;
      float vA = acc0[tl][r] * cs + fb;
      float vB = acc1[tl][r] * cs + fb;
      lmax = fmaxf(lmax, fmaxf(fabsf(vA), fabsf(vB)));
      out[(size_t)mA * EDIM + e] = vA;
      out[(size_t)mB * EDIM + e] = vB;
    }
  }
  // wave32 reduce then block reduce then device atomic (uint bits: valid for >=0)
  for (int off = 16; off >= 1; off >>= 1)
    lmax = fmaxf(lmax, __shfl_xor(lmax, off, 32));
  if (lane == 0) wmax[wave] = lmax;
  __syncthreads();
  if (t == 0) {
    float bm = wmax[0];
#pragma unroll
    for (int i = 1; i < 8; ++i) bm = fmaxf(bm, wmax[i]);
    atomicMax(actmax, __float_as_uint(bm));
  }
}

// ---------------------------------------------------------------------------
// Kernel 3: requantize in place: round(out/act_sf)*act_sf; append act_sf.
// ---------------------------------------------------------------------------
__global__ __launch_bounds__(256) void requant_kernel(
    float* __restrict__ out, const unsigned* __restrict__ actmax,
    float* __restrict__ sf_out) {
  const size_t i = (size_t)blockIdx.x * 256 + threadIdx.x;
  const float amax = __uint_as_float(*actmax);
  const float sf   = amax * (1.0f / QMAXF);
  const float inv  = QMAXF / amax;
  float4* p = (float4*)out;
  float4 v = p[i];
  v.x = rintf(v.x * inv) * sf;
  v.y = rintf(v.y * inv) * sf;
  v.z = rintf(v.z * inv) * sf;
  v.w = rintf(v.w * inv) * sf;
  p[i] = v;
  if (i == 0) *sf_out = sf;
}

// ---------------------------------------------------------------------------
extern "C" void kernel_launch(void* const* d_in, const int* in_sizes, int n_in,
                              void* d_out, int out_size, void* d_ws,
                              size_t ws_size, hipStream_t stream) {
  (void)in_sizes; (void)n_in; (void)out_size; (void)ws_size;
  const float* x    = (const float*)d_in[0];   // [128,3,224,224]
  const float* w    = (const float*)d_in[1];   // [768,3,16,16]
  const float* bias = (const float*)d_in[2];   // [768]
  const float* sf   = (const float*)d_in[3];   // [1]

  char* ws = (char*)d_ws;
  unsigned* actmax = (unsigned*)ws;                                   // 4 B
  _Float16* wq     = (_Float16*)(ws + 64);                            // 768*768*2
  float* conv_sf   = (float*)(ws + 64 + EDIM * KDIM * 2);             // 768*4
  float* fbias     = (float*)(ws + 64 + EDIM * KDIM * 2 + EDIM * 4);  // 768*4

  float* out    = (float*)d_out;                // [25088, 768]
  float* sf_out = (float*)d_out + (size_t)MTOT * EDIM;

  hipMemsetAsync(d_ws, 0, 64, stream);  // zero the absmax scalar every call

  wquant_kernel<<<EDIM, 256, 0, stream>>>(w, bias, sf, wq, conv_sf, fbias);
  gemm_kernel<<<MTOT / MTILE, 256, 0, stream>>>(x, wq, conv_sf, fbias, sf, out,
                                                actmax);
  requant_kernel<<<(MTOT * EDIM) / (256 * 4), 256, 0, stream>>>(out, actmax,
                                                                sf_out);
}
